// Reservoir_84327387889916
// MI455X (gfx1250) — matrix operand order repaired and checked by reading, not compiled, not run on hardware
//
#include <hip/hip_runtime.h>
#include <hip/hip_bf16.h>
#include <math.h>

#define T_STEPS 8192
#define IN_F    128
#define SZ      4096
#define OUT_F   64
#define NBLK    128                // persistent blocks for the scan
#define ROWS_PER_BLK (SZ / NBLK)   // 32 rows per block
#define TPR     8                  // threads per row
#define SEGLEN  (SZ / TPR)         // 512-long partial dot per thread

typedef __attribute__((ext_vector_type(16))) __bf16 v16bf;
typedef __attribute__((ext_vector_type(8)))  float  v8f;
typedef __attribute__((ext_vector_type(4)))  unsigned int uint32x4;
typedef __attribute__((ext_vector_type(8)))  int   int32x8;
typedef __attribute__((ext_vector_type(4)))  int   int32x4;

// ---------------- workspace layout (bytes) ----------------
static const size_t OFF_DRIVE = 0;                                        // f32 [T][SZ]
static const size_t OFF_UBF   = (size_t)T_STEPS * SZ * 4;                 // bf16 [T][IN]
static const size_t OFF_WINBF = OFF_UBF   + (size_t)T_STEPS * IN_F * 2;   // bf16 [SZ][IN]
static const size_t OFF_WBF   = OFF_WINBF + (size_t)SZ * IN_F * 2;        // bf16 [SZ][SZ]
static const size_t OFF_XBUF  = OFF_WBF   + (size_t)SZ * SZ * 2;          // f32 [2][SZ]
static const size_t OFF_SYNC  = OFF_XBUF  + (size_t)2 * SZ * 4;           // 2 x u32

// float -> bf16 (round-to-nearest-even), no reliance on __bf16 arithmetic
static __device__ inline __bf16 f2bf(float f) {
    unsigned u = __float_as_uint(f);
    unsigned r = (u + 0x7fffu + ((u >> 16) & 1u)) >> 16;
    unsigned short s = (unsigned short)r;
    __bf16 o;
    __builtin_memcpy(&o, &s, 2);
    return o;
}
static __device__ inline float bf_lo(unsigned p) { return __uint_as_float(p << 16); }
static __device__ inline float bf_hi(unsigned p) { return __uint_as_float(p & 0xffff0000u); }

// ---------------- init: zero barrier state, x0 = X ----------------
__global__ void init_kernel(const float* __restrict__ X, float* __restrict__ xbuf,
                            unsigned* __restrict__ sync) {
    int i = blockIdx.x * 256 + threadIdx.x;
    if (i < SZ) xbuf[i] = X[i];
    if (i < 2)  sync[i] = 0u;
}

// ---------------- convert u, win, w to bf16 ----------------
__global__ void convert_kernel(const float* __restrict__ u, const float* __restrict__ win,
                               const float* __restrict__ w, __bf16* __restrict__ ubf,
                               __bf16* __restrict__ winbf, __bf16* __restrict__ wbf) {
    size_t i = (size_t)blockIdx.x * 256 + threadIdx.x;
    const size_t NU = (size_t)T_STEPS * IN_F;
    const size_t NW = (size_t)SZ * IN_F;
    const size_t NWW = (size_t)SZ * SZ;
    if (i < NU)                 ubf[i]            = f2bf(u[i]);
    else if (i < NU + NW)       winbf[i - NU]     = f2bf(win[i - NU]);
    else if (i < NU + NW + NWW) wbf[i - NU - NW]  = f2bf(w[i - NU - NW]);
}

// ---------------- drive = u @ win^T via v_wmma_f32_16x16x32_bf16 ----------------
__global__ __launch_bounds__(256) void drive_wmma_kernel(const __bf16* __restrict__ ubf,
                                                         const __bf16* __restrict__ winbf,
                                                         float* __restrict__ drive) {
    const int wave = threadIdx.x >> 5;
    const int lane = threadIdx.x & 31;
    const int tile = blockIdx.x * 8 + wave;       // (T/16)*(SZ/16) tiles total
    const int tN   = tile & (SZ / 16 - 1);        // 256 tiles along SZ
    const int tM   = tile >> 8;
    const int half = lane >> 4;
    const int l15  = lane & 15;

    const __bf16* arow = ubf   + (size_t)(tM * 16 + l15) * IN_F;
    const __bf16* brow = winbf + (size_t)(tN * 16 + l15) * IN_F;

    v8f c = {};
#pragma unroll
    for (int k0 = 0; k0 < IN_F; k0 += 32) {
        v16bf a, b;
        const __bf16* pa0 = arow + k0 + half * 8;
        const __bf16* pa1 = arow + k0 + 16 + half * 8;
        const __bf16* pb0 = brow + k0 + half * 8;
        const __bf16* pb1 = brow + k0 + 16 + half * 8;
#pragma unroll
        for (int i = 0; i < 8; ++i) { a[i] = pa0[i]; a[8 + i] = pa1[i]; }
#pragma unroll
        for (int i = 0; i < 8; ++i) { b[i] = pb0[i]; b[8 + i] = pb1[i]; }
        c = __builtin_amdgcn_wmma_f32_16x16x32_bf16(false, a, false, b,
                                                    (short)0, c, false, false);
    }
    float* dp = drive + (size_t)(tM * 16 + half * 8) * SZ + tN * 16 + l15;
#pragma unroll
    for (int j = 0; j < 8; ++j) dp[(size_t)j * SZ] = c[j];
}

// ---------------- TDM: DMA one 16KB row (x_t) from global into LDS ----------------
// 2-D descriptor: data_size=4B, tensor 4096x1, tile 4096x1 (contiguous copy).
static __device__ inline void tdm_load_x_to_lds(float* xs, const float* xsrc) {
#if __has_builtin(__builtin_amdgcn_tensor_load_to_lds)
    unsigned lds_off = (unsigned)(size_t)(void*)xs;         // flat LDS addr truncates to offset
    unsigned long long ga = (unsigned long long)(size_t)xsrc;
    uint32x4 g0;
    g0[0] = 0x1u;                               // count=1, is_restore=0, gather off
    g0[1] = lds_off;                            // lds_addr
    g0[2] = (unsigned)(ga & 0xffffffffu);       // global_addr[31:0]
    g0[3] = (unsigned)((ga >> 32) & 0x1ffffffu) // global_addr[56:32]
          | 0x80000000u;                        // type = 2 ("image")
    int32x8 g1;
    g1[0] = 0x00020000;                         // wg_mask=0, data_size=2 (4B)
    g1[1] = (int)(0x10000000u);                 // tensor_dim0[15:0]=4096 in [31:16]
    g1[2] = (int)(0x00010000u);                 // tensor_dim1[15:0]=1 in [31:16]
    g1[3] = (int)(0x10000000u);                 // tile_dim0=4096 in [31:16]
    g1[4] = 0x00000001;                         // tile_dim1=1, tile_dim2=0
    g1[5] = 0x00001000;                         // tensor_dim0_stride=4096
    g1[6] = (int)(0x10000000u);                 // dim1_stride[15:0]=4096 in [31:16]
    g1[7] = 0;
    int32x4 z4 = {0, 0, 0, 0};
#if __clang_major__ >= 23
    int32x8 z8 = {0, 0, 0, 0, 0, 0, 0, 0};
    __builtin_amdgcn_tensor_load_to_lds(g0, g1, z4, z4, z8, 0);
#else
    __builtin_amdgcn_tensor_load_to_lds(g0, g1, z4, z4, 0);
#endif
    __builtin_amdgcn_s_wait_tensorcnt(0);
#else
    // fallback: cooperative vector loads (single wave)
    for (int i = (int)(threadIdx.x & 31); i < SZ / 4; i += 32)
        ((float4*)xs)[i] = ((const float4*)xsrc)[i];
#endif
}

// ---------------- persistent scan with device-wide barrier ----------------
__global__ __launch_bounds__(256) void scan_kernel(const unsigned* __restrict__ wbf2, // bf16 pairs [SZ][SZ/2]
                                                   const float* __restrict__ drive,
                                                   const float* __restrict__ llw,
                                                   const float* __restrict__ llb,
                                                   float* __restrict__ xbuf,   // [2][SZ]
                                                   unsigned* __restrict__ sync,
                                                   float* __restrict__ out) {  // [T][OUT] ++ x_final[SZ]
    __shared__ float xs[SZ];     // 16 KB state cache, TDM-filled each step
    __shared__ float red[256];

    const int tid  = threadIdx.x;
    const int bid  = blockIdx.x;
    const int rowl = tid >> 3;                    // 0..31
    const int q    = tid & (TPR - 1);             // eighth of the dot
    const int row  = bid * ROWS_PER_BLK + rowl;

    int cur = 0;
    for (int t = 0; t < T_STEPS; ++t) {
        // stage x_t in LDS via the Tensor Data Mover (one DMA per block)
        if (tid < 32)
            tdm_load_x_to_lds(xs, xbuf + (size_t)cur * SZ);
        __syncthreads();

        // partial dot over 512 weights (256 bf16 pairs = 64 uint4)
        const uint4* wp4 = (const uint4*)(wbf2 + (size_t)row * (SZ / 2) + q * (SEGLEN / 2));
        const float* xp  = xs + q * SEGLEN;
        float acc = 0.f;
#pragma unroll 4
        for (int j = 0; j < SEGLEN / 8; ++j) {
            uint4 wv = wp4[j];
            const float* xq = xp + j * 8;
            acc = fmaf(bf_lo(wv.x), xq[0], acc); acc = fmaf(bf_hi(wv.x), xq[1], acc);
            acc = fmaf(bf_lo(wv.y), xq[2], acc); acc = fmaf(bf_hi(wv.y), xq[3], acc);
            acc = fmaf(bf_lo(wv.z), xq[4], acc); acc = fmaf(bf_hi(wv.z), xq[5], acc);
            acc = fmaf(bf_lo(wv.w), xq[6], acc); acc = fmaf(bf_hi(wv.w), xq[7], acc);
        }
        red[tid] = acc;
        __syncthreads();
        if (q == 0) {
            float dot = 0.f;
#pragma unroll
            for (int k = 0; k < TPR; ++k) dot += red[tid + k];
            float xn = tanhf(drive[(size_t)t * SZ + row] + dot);
            xbuf[(size_t)(cur ^ 1) * SZ + row] = xn;
            if (t + 1 < T_STEPS)
                __builtin_prefetch(&drive[(size_t)(t + 1) * SZ + row], 0, 0);
        }

        // ---- device-wide barrier (monotonic counter, release/acquire) ----
        __threadfence();            // make this wave's x_next stores visible
        __syncthreads();
        if (tid == 0) {
            unsigned prev = atomicAdd(sync, 1u);
            unsigned target = (unsigned)(t + 1) * NBLK;
            if (prev + 1u == target) {
                __threadfence();
                atomicExch(sync + 1, (unsigned)(t + 1));   // release
            } else {
                while (atomicAdd(sync + 1, 0u) < (unsigned)(t + 1))
                    __builtin_amdgcn_s_sleep(1);
            }
        }
        __syncthreads();
        __threadfence();            // acquire: see other blocks' x_next
        cur ^= 1;

        // ---- fused readout p[t] = ll_w @ states[t] + b  (blocks 0..15) ----
        if (bid < 16) {
            const int o  = bid * 4 + (tid >> 6);   // output feature
            const int ck = tid & 63;               // 64-way split of the 4096-dot
            const float* xr = xbuf + (size_t)cur * SZ + ck * 64;
            const float* lw = llw + (size_t)o * SZ + ck * 64;
            float pa = 0.f;
#pragma unroll 8
            for (int k = 0; k < 64; ++k) pa = fmaf(lw[k], xr[k], pa);
            red[tid] = pa;
            __syncthreads();
            if (ck == 0) {
                float s = 0.f;
#pragma unroll
                for (int k = 0; k < 64; ++k) s += red[tid + k];
                out[(size_t)t * OUT_F + o] = s + llb[o];
            }
            __syncthreads();
        }
    }

    // x_final = states[T-1]
    int gi = bid * 256 + tid;
    if (gi < SZ) out[(size_t)T_STEPS * OUT_F + gi] = xbuf[(size_t)cur * SZ + gi];
}

extern "C" void kernel_launch(void* const* d_in, const int* in_sizes, int n_in,
                              void* d_out, int out_size, void* d_ws, size_t ws_size,
                              hipStream_t stream) {
    (void)in_sizes; (void)n_in; (void)out_size; (void)ws_size;
    const float* u    = (const float*)d_in[0];   // [T, IN]
    const float* X    = (const float*)d_in[1];   // [SZ]
    const float* win  = (const float*)d_in[2];   // [SZ, IN]
    const float* w    = (const float*)d_in[3];   // [SZ, SZ]
    const float* ll_w = (const float*)d_in[4];   // [OUT, SZ]
    const float* ll_b = (const float*)d_in[5];   // [OUT]
    float* out = (float*)d_out;

    char* ws = (char*)d_ws;
    float*    drive = (float*)   (ws + OFF_DRIVE);
    __bf16*   ubf   = (__bf16*)  (ws + OFF_UBF);
    __bf16*   winbf = (__bf16*)  (ws + OFF_WINBF);
    __bf16*   wbf   = (__bf16*)  (ws + OFF_WBF);
    float*    xbuf  = (float*)   (ws + OFF_XBUF);
    unsigned* syncp = (unsigned*)(ws + OFF_SYNC);

    // 1) init barrier + x0
    init_kernel<<<(SZ + 255) / 256, 256, 0, stream>>>(X, xbuf, syncp);

    // 2) fp32 -> bf16 for u, win, w
    const size_t ncvt = (size_t)T_STEPS * IN_F + (size_t)SZ * IN_F + (size_t)SZ * SZ;
    convert_kernel<<<(unsigned)((ncvt + 255) / 256), 256, 0, stream>>>(u, win, w, ubf, winbf, wbf);

    // 3) drive GEMM: (T/16)*(SZ/16) tiles, 8 waves (tiles) per 256-thread block
    const unsigned ntiles = (T_STEPS / 16) * (SZ / 16);
    drive_wmma_kernel<<<ntiles / 8, 256, 0, stream>>>(ubf, winbf, drive);

    // 4) sequential scan + fused readout, persistent grid
    scan_kernel<<<NBLK, 256, 0, stream>>>((const unsigned*)wbf, drive, ll_w, ll_b,
                                          xbuf, syncp, out);
}
